// CasualMultiHeadAttention_82566451298978
// MI455X (gfx1250) — matrix operand order repaired and checked by reading, not compiled, not run on hardware
//
#include <hip/hip_runtime.h>

#define D_MODEL 1024
#define NUM_HEADS 16
#define DK 64
#define SEQ 2048
#define BATCH 2

typedef __attribute__((ext_vector_type(16))) _Float16 v16h;
typedef __attribute__((ext_vector_type(8)))  float    v8f;

union FragH { unsigned int u[8]; v16h v; };

__device__ __forceinline__ unsigned int pkrtz(float lo, float hi) {
    return __builtin_bit_cast(unsigned int, __builtin_amdgcn_cvt_pkrtz(lo, hi));
}
__device__ __forceinline__ unsigned ldsoff(const void* p) {
    return (unsigned)(size_t)p;          // generic LDS ptr low 32 bits = LDS offset
}
__device__ __forceinline__ void async_b128(unsigned lds, unsigned long long gaddr) {
    asm volatile("global_load_async_to_lds_b128 %0, %1, off"
                 :: "v"(lds), "v"(gaddr) : "memory");
}
template<int N>
__device__ __forceinline__ void wait_async() {
    asm volatile("s_wait_asynccnt %0" :: "i"(N) : "memory");
}

// ---------------------------------------------------------------------------
// fp32 -> f16 conversion pass (one v_cvt_pk_rtz_f16_f32 per pair)
// ---------------------------------------------------------------------------
__global__ __launch_bounds__(256)
void cvt_f16_kernel(const float* __restrict__ src, unsigned int* __restrict__ dst, int nPairs)
{
    int i = blockIdx.x * 256 + threadIdx.x;
    if (i >= nPairs) return;
    float2 v = ((const float2*)src)[i];
    dst[i] = pkrtz(v.x, v.y);
}

// ---------------------------------------------------------------------------
// Y = X * W^T  (X: R x K f16, W: N x K f16)
// out_mode 0: fp32 row-major R x N  (final projection -> d_out)
// out_mode 1: f16 into (B,H,S,dk)   (Q/K/V projections)
// block = 128 threads (4 waves), tile 128x64, K-step 32, f16 WMMA.
// Double-buffered LDS staging via global_load_async_to_lds_b128:
// issue tile kt+1, wait ASYNCcnt down to the new group, compute tile kt.
// ---------------------------------------------------------------------------
__global__ __launch_bounds__(128)
void gemm_xwT_kernel(const unsigned short* __restrict__ X, const unsigned short* __restrict__ W,
                     float* __restrict__ outF, _Float16* __restrict__ outH,
                     int R, int N, int Kd, int out_mode, int S, int H, int dk)
{
    __shared__ unsigned short Xs[2][128 * 32];   // 8 KB per buffer
    __shared__ unsigned short Ws[2][64 * 32];    // 4 KB per buffer

    const int tid  = threadIdx.x;
    const int lane = tid & 31;
    const int wave = tid >> 5;
    const int tileM = blockIdx.x * 128;
    const int tileN = blockIdx.y * 64;

    auto stage = [&](int k0, int buf) {
        #pragma unroll
        for (int q = 0; q < 4; ++q) {            // X: 512 b128 chunks
            int chunk = q * 128 + tid;
            int row = chunk >> 2;                // 0..127
            int kc  = (chunk & 3) * 8;
            async_b128(ldsoff(&Xs[buf][row * 32 + kc]),
                       (unsigned long long)(size_t)(X + (size_t)(tileM + row) * Kd + k0 + kc));
        }
        #pragma unroll
        for (int q = 0; q < 2; ++q) {            // W: 256 b128 chunks
            int chunk = q * 128 + tid;
            int row = chunk >> 2;                // 0..63
            int kc  = (chunk & 3) * 8;
            async_b128(ldsoff(&Ws[buf][row * 32 + kc]),
                       (unsigned long long)(size_t)(W + (size_t)(tileN + row) * Kd + k0 + kc));
        }
        __builtin_prefetch(X + (size_t)(tileM + (tid & 127)) * Kd + k0 + 32, 0, 0);
    };

    v8f acc[2][4] = {};
    const int nK = Kd / 32;
    stage(0, 0);

    for (int kt = 0; kt < nK; ++kt) {
        const int cur = kt & 1, nxt = cur ^ 1;
        const bool more = (kt + 1 < nK);
        if (more) { stage((kt + 1) * 32, nxt); wait_async<6>(); }
        else      { wait_async<0>(); }
        __syncthreads();

        // two A fragments per wave (rows wave*32 + {0,16})
        FragH a[2];
        #pragma unroll
        for (int rr = 0; rr < 2; ++rr) {
            int row = wave * 32 + rr * 16 + (lane & 15);
            #pragma unroll
            for (int v = 0; v < 8; ++v) {
                int k = ((v < 4) ? 0 : 16) + ((lane < 16) ? 0 : 8) + 2 * (v & 3);
                a[rr].u[v] = *(const unsigned int*)&Xs[cur][row * 32 + k];
            }
        }
        // 4 B column tiles, each reused across both A fragments: 8 WMMAs
        #pragma unroll
        for (int c = 0; c < 4; ++c) {
            FragH bf;
            int n = c * 16 + (lane & 15);
            #pragma unroll
            for (int v = 0; v < 8; ++v) {
                int k = ((v < 4) ? 0 : 16) + ((lane < 16) ? 0 : 8) + 2 * (v & 3);
                bf.u[v] = *(const unsigned int*)&Ws[cur][n * 32 + k];
            }
            #pragma unroll
            for (int rr = 0; rr < 2; ++rr)
                acc[rr][c] = __builtin_amdgcn_wmma_f32_16x16x32_f16(
                    false, a[rr].v, false, bf.v, (short)0, acc[rr][c], false, false);
        }
        __syncthreads();
    }

    // store: C layout -> row = vgpr + (lane<16 ? 0 : 8), col = lane%16
    #pragma unroll
    for (int rr = 0; rr < 2; ++rr)
    #pragma unroll
    for (int c = 0; c < 4; ++c)
        #pragma unroll
        for (int r = 0; r < 8; ++r) {
            int row = tileM + wave * 32 + rr * 16 + r + ((lane < 16) ? 0 : 8);
            int col = tileN + c * 16 + (lane & 15);
            float val = acc[rr][c][r];
            if (out_mode == 0) {
                outF[(size_t)row * N + col] = val;
            } else {
                int b = row / S, s = row % S;
                int h = col / dk, dd = col % dk;
                outH[(((size_t)(b * H + h) * S + s) * dk) + dd] = (_Float16)val;
            }
        }
}

// ---------------------------------------------------------------------------
// In-place RoPE on f16 Q and K, layout (B,H,S,dk)
// ---------------------------------------------------------------------------
__global__ __launch_bounds__(256)
void rope_kernel(_Float16* __restrict__ Qh, _Float16* __restrict__ Kh,
                 const int* __restrict__ pos, int total)
{
    int idx = blockIdx.x * 256 + threadIdx.x;
    if (idx >= total) return;
    int j  = idx & 31;                 // dk/2 = 32 pairs
    int s  = (idx >> 5) & (SEQ - 1);
    int bh = idx >> 16;                // 5 + 11 bits
    size_t base = ((size_t)bh * SEQ + s) * DK + 2 * j;

    float p   = (float)pos[s];
    float inv = __powf(10000.0f, -(2.0f * (float)j) / 64.0f);
    float sn, cs;
    __sincosf(p * inv, &sn, &cs);
    {
        float e = (float)Qh[base], o = (float)Qh[base + 1];
        Qh[base]     = (_Float16)(e * cs - o * sn);
        Qh[base + 1] = (_Float16)(e * sn + o * cs);
    }
    {
        float e = (float)Kh[base], o = (float)Kh[base + 1];
        Kh[base]     = (_Float16)(e * cs - o * sn);
        Kh[base + 1] = (_Float16)(e * sn + o * cs);
    }
}

// ---------------------------------------------------------------------------
// Flash attention: block = (b, h, 64 query rows), 4 waves x 16 rows each.
// 32-key steps, double-buffered: K tile async-staged (ping-pong), V tile
// pipelined through registers and transposed into LDS after compute.
// Online softmax with per-wave LDS scratch; f16 output in (B,S,D).
// ---------------------------------------------------------------------------
__global__ __launch_bounds__(128)
void flash_attn_kernel(const unsigned short* __restrict__ Qh,
                       const unsigned short* __restrict__ Kh,
                       const unsigned short* __restrict__ Vh,
                       _Float16* __restrict__ O)
{
    __shared__ unsigned short Ks[2][32 * 64];     // [key][dk]
    __shared__ unsigned short Vt[2][64 * 32];     // [dk][key]
    __shared__ float          Sc[4][16 * 32];     // per-wave score scratch
    __shared__ unsigned short Pw[4][16 * 32];     // per-wave probs (f16)
    __shared__ float          Alph[4][16];
    __shared__ float          Lb[4][16];

    const int tid  = threadIdx.x;
    const int lane = tid & 31;
    const int wave = tid >> 5;
    const int qt   = blockIdx.x;                  // 0..31
    const int h    = blockIdx.y;
    const int b    = blockIdx.z;
    const int q0w  = qt * 64 + wave * 16;
    const size_t headBase = (size_t)(b * NUM_HEADS + h) * SEQ * DK;
    const float scale = 0.125f;                   // 1/sqrt(64)

    unsigned vr[8];                               // V-tile pipeline registers

    auto issueK = [&](int kb, int buf) {
        #pragma unroll
        for (int q = 0; q < 2; ++q) {             // 256 b128 chunks
            int chunk = q * 128 + tid;
            int key = chunk >> 3;
            int kc  = (chunk & 7) * 8;
            async_b128(ldsoff(&Ks[buf][key * 64 + kc]),
                       (unsigned long long)(size_t)(Kh + headBase + (size_t)(kb + key) * DK + kc));
        }
    };
    auto loadV = [&](int kb) {
        #pragma unroll
        for (int q = 0; q < 8; ++q) {
            int slot = q * 128 + tid;             // 0..1023 u32 slots
            int key  = slot >> 5;
            int du   = slot & 31;
            vr[q] = *(const unsigned int*)(Vh + headBase + (size_t)(kb + key) * DK + du * 2);
        }
    };
    auto storeV = [&](int buf) {
        #pragma unroll
        for (int q = 0; q < 8; ++q) {
            int slot = q * 128 + tid;
            int key  = slot >> 5;
            int du   = slot & 31;
            Vt[buf][(du * 2    ) * 32 + key] = (unsigned short)(vr[q] & 0xFFFFu);
            Vt[buf][(du * 2 + 1) * 32 + key] = (unsigned short)(vr[q] >> 16);
        }
    };

    // preload Q fragments (dk halves 0-31, 32-63)
    FragH qf[2];
    {
        const unsigned short* qp = Qh + headBase + (size_t)(q0w + (lane & 15)) * DK;
        #pragma unroll
        for (int kk = 0; kk < 2; ++kk)
            #pragma unroll
            for (int v = 0; v < 8; ++v) {
                int k = kk * 32 + ((v < 4) ? 0 : 16) + ((lane < 16) ? 0 : 8) + 2 * (v & 3);
                qf[kk].u[v] = *(const unsigned int*)(qp + k);
            }
    }

    v8f o[4] = {};
    float m = -1e30f, l = 0.0f;                   // row state in lanes 0..15

    const int nSteps = qt * 2 + 2;                // causal bound for this block

    // prologue: stage tile 0
    issueK(0, 0);
    loadV(0);
    storeV(0);

    for (int jt = 0; jt < nSteps; ++jt) {
        const int cur = jt & 1, nxt = cur ^ 1;
        const int kb = jt * 32;
        const bool more = (jt + 1 < nSteps);
        if (more) { issueK(kb + 32, nxt); loadV(kb + 32); wait_async<2>(); }
        else      { wait_async<0>(); }
        __syncthreads();

        // S = Q K^T: 2 key-column tiles x 2 dk halves
        #pragma unroll
        for (int c = 0; c < 2; ++c) {
            v8f sacc = {};
            #pragma unroll
            for (int kk = 0; kk < 2; ++kk) {
                FragH bf;
                int n = c * 16 + (lane & 15);     // key within tile
                #pragma unroll
                for (int v = 0; v < 8; ++v) {
                    int k = kk * 32 + ((v < 4) ? 0 : 16) + ((lane < 16) ? 0 : 8) + 2 * (v & 3);   // dk
                    bf.u[v] = *(const unsigned int*)&Ks[cur][n * 64 + k];
                }
                sacc = __builtin_amdgcn_wmma_f32_16x16x32_f16(
                    false, qf[kk].v, false, bf.v, (short)0, sacc, false, false);
            }
            // spill scores with causal mask
            #pragma unroll
            for (int r = 0; r < 8; ++r) {
                int row = r + ((lane < 16) ? 0 : 8);
                int col = c * 16 + (lane & 15);
                float sv = sacc[r] * scale;
                if (kb + col > q0w + row) sv = -1e9f;
                Sc[wave][row * 32 + col] = sv;
            }
        }
        __builtin_amdgcn_wave_barrier();

        // online softmax: lane r (0..15) owns query row r
        if (lane < 16) {
            int r = lane;
            float mx = m;
            #pragma unroll
            for (int i = 0; i < 32; ++i) mx = fmaxf(mx, Sc[wave][r * 32 + i]);
            float al = __expf(m - mx);
            float sum = 0.0f;
            #pragma unroll
            for (int i = 0; i < 16; ++i) {
                float p0 = __expf(Sc[wave][r * 32 + 2 * i]     - mx);
                float p1 = __expf(Sc[wave][r * 32 + 2 * i + 1] - mx);
                sum += p0 + p1;
                *(unsigned int*)&Pw[wave][r * 32 + 2 * i] = pkrtz(p0, p1);
            }
            m = mx;
            l = l * al + sum;
            Alph[wave][r] = al;
        }
        __builtin_amdgcn_wave_barrier();

        // rescale O accumulators
        #pragma unroll
        for (int r = 0; r < 8; ++r) {
            int row = r + ((lane < 16) ? 0 : 8);
            float a = Alph[wave][row];
            o[0][r] *= a; o[1][r] *= a; o[2][r] *= a; o[3][r] *= a;
        }

        // P fragment (16 rows x 32 keys)
        FragH pf;
        {
            int row = lane & 15;
            #pragma unroll
            for (int v = 0; v < 8; ++v) {
                int k = ((v < 4) ? 0 : 16) + ((lane < 16) ? 0 : 8) + 2 * (v & 3);
                pf.u[v] = *(const unsigned int*)&Pw[wave][row * 32 + k];
            }
        }
        // O += P V : 4 dk-column tiles
        #pragma unroll
        for (int t = 0; t < 4; ++t) {
            FragH vf;
            int n = t * 16 + (lane & 15);         // dk col
            #pragma unroll
            for (int v = 0; v < 8; ++v) {
                int k = ((v < 4) ? 0 : 16) + ((lane < 16) ? 0 : 8) + 2 * (v & 3);   // key
                vf.u[v] = *(const unsigned int*)&Vt[cur][n * 32 + k];
            }
            o[t] = __builtin_amdgcn_wmma_f32_16x16x32_f16(
                false, pf.v, false, vf.v, (short)0, o[t], false, false);
        }
        __syncthreads();

        if (more) storeV(nxt);   // transpose next V tile into the idle buffer
    }

    if (lane < 16) Lb[wave][lane] = l;
    __builtin_amdgcn_wave_barrier();

    #pragma unroll
    for (int r = 0; r < 8; ++r) {
        int row = r + ((lane < 16) ? 0 : 8);
        float linv = 1.0f / Lb[wave][row];
        int srow = q0w + row;
        size_t ob = ((size_t)b * SEQ + srow) * D_MODEL + h * DK + (lane & 15);
        O[ob +  0] = (_Float16)(o[0][r] * linv);
        O[ob + 16] = (_Float16)(o[1][r] * linv);
        O[ob + 32] = (_Float16)(o[2][r] * linv);
        O[ob + 48] = (_Float16)(o[3][r] * linv);
    }
}

// ---------------------------------------------------------------------------
extern "C" void kernel_launch(void* const* d_in, const int* in_sizes, int n_in,
                              void* d_out, int out_size, void* d_ws, size_t ws_size,
                              hipStream_t stream)
{
    const float* x   = (const float*)d_in[0];
    const float* Wq  = (const float*)d_in[1];
    const float* Wk  = (const float*)d_in[2];
    const float* Wv  = (const float*)d_in[3];
    const float* Wo  = (const float*)d_in[4];
    const int*   pos = (const int*)d_in[5];

    char* ws = (char*)d_ws;
    const size_t MB = 1024 * 1024;
    unsigned short* xh    = (unsigned short*)(ws);            // 8 MB
    unsigned short* Wqh   = (unsigned short*)(ws +  8 * MB);  // 2 MB
    unsigned short* Wkh   = (unsigned short*)(ws + 10 * MB);
    unsigned short* Wvh   = (unsigned short*)(ws + 12 * MB);
    unsigned short* Woh   = (unsigned short*)(ws + 14 * MB);
    unsigned short* Qhb   = (unsigned short*)(ws + 16 * MB);  // 8 MB
    unsigned short* Khb   = (unsigned short*)(ws + 24 * MB);
    unsigned short* Vhb   = (unsigned short*)(ws + 32 * MB);
    unsigned short* AttnH = (unsigned short*)(ws + 40 * MB);  // 8 MB
    float* out = (float*)d_out;

    const int R = BATCH * SEQ;                       // 4096
    const int xPairs = R * D_MODEL / 2;              // 2^21
    const int wPairs = D_MODEL * D_MODEL / 2;        // 2^19

    cvt_f16_kernel<<<(xPairs + 255) / 256, 256, 0, stream>>>(x,  (unsigned int*)xh,  xPairs);
    cvt_f16_kernel<<<(wPairs + 255) / 256, 256, 0, stream>>>(Wq, (unsigned int*)Wqh, wPairs);
    cvt_f16_kernel<<<(wPairs + 255) / 256, 256, 0, stream>>>(Wk, (unsigned int*)Wkh, wPairs);
    cvt_f16_kernel<<<(wPairs + 255) / 256, 256, 0, stream>>>(Wv, (unsigned int*)Wvh, wPairs);
    cvt_f16_kernel<<<(wPairs + 255) / 256, 256, 0, stream>>>(Wo, (unsigned int*)Woh, wPairs);

    dim3 gGemm(R / 128, D_MODEL / 64);               // 32 x 16
    gemm_xwT_kernel<<<gGemm, 128, 0, stream>>>(xh, Wqh, nullptr, (_Float16*)Qhb, R, D_MODEL, D_MODEL, 1, SEQ, NUM_HEADS, DK);
    gemm_xwT_kernel<<<gGemm, 128, 0, stream>>>(xh, Wkh, nullptr, (_Float16*)Khb, R, D_MODEL, D_MODEL, 1, SEQ, NUM_HEADS, DK);
    gemm_xwT_kernel<<<gGemm, 128, 0, stream>>>(xh, Wvh, nullptr, (_Float16*)Vhb, R, D_MODEL, D_MODEL, 1, SEQ, NUM_HEADS, DK);

    int total = BATCH * NUM_HEADS * SEQ * (DK / 2);  // 2^21
    rope_kernel<<<(total + 255) / 256, 256, 0, stream>>>((_Float16*)Qhb, (_Float16*)Khb, pos, total);

    dim3 gF(SEQ / 64, NUM_HEADS, BATCH);
    flash_attn_kernel<<<gF, 128, 0, stream>>>(Qhb, Khb, Vhb, (_Float16*)AttnH);

    gemm_xwT_kernel<<<gGemm, 128, 0, stream>>>(AttnH, Woh, out, nullptr, R, D_MODEL, D_MODEL, 0, SEQ, NUM_HEADS, DK);
}
